// GAT_57440892616777
// MI455X (gfx1250) — matrix operand order repaired and checked by reading, not compiled, not run on hardware
//
#include <hip/hip_runtime.h>
#include <math.h>

constexpr int N_NODES = 100000;
constexpr int N_EDGES = 1600000;
constexpr int E_TOT   = N_EDGES + N_NODES;   // PyG self-loops appended
constexpr int F_IN    = 256;
constexpr int HID     = 128;                 // HEADS*HIDDEN
constexpr int HEADS   = 4;
constexpr int HDIM    = 32;
constexpr int NC      = 16;
constexpr float NEG_SLOPE = 0.2f;

typedef __attribute__((ext_vector_type(16))) __bf16 v16bf;
typedef __attribute__((ext_vector_type(8)))  float  v8f;

__device__ inline v8f wmma_bf16(v16bf a, v16bf b, v8f c) {
  return __builtin_amdgcn_wmma_f32_16x16x32_bf16(false, a, false, b, (short)0, c, false, false);
}

struct bf2 { __bf16 hi, lo; };
__device__ inline bf2 split_bf16(float f) {
  bf2 r;
  r.hi = (__bf16)f;
  r.lo = (__bf16)(f - (float)r.hi);
  return r;
}

__device__ inline void atomicMaxF(float* addr, float val) {
  if (val >= 0.0f) atomicMax((int*)addr, __float_as_int(val));
  else             atomicMin((unsigned int*)addr, __float_as_uint(val));
}

// ---------------- GEMM1: h1[100000,128] = x[100000,256] @ W1[256,128] (bf16x3 WMMA) ----
__global__ void __launch_bounds__(256)
k_gemm1(const float* __restrict__ x, const float* __restrict__ W, float* __restrict__ h1) {
  const int wave = threadIdx.x >> 5;
  const int lane = threadIdx.x & 31;
  const int m0 = blockIdx.x * 16;      // 6250 blocks, exact
  const int n0 = wave * 16;            // 8 waves cover N=128

  const int am  = lane & 15;           // A: row
  const int akb = (lane >> 4) * 16;    // A: K base (0 or 16)
  const int bn  = lane & 15;           // B: col
  const int bkb = (lane >> 4) * 16;    // B: K base (0 or 16)

  const float* xrow = x + (size_t)(m0 + am) * F_IN;
  const float* wcol = W + (n0 + bn);

  v8f c = {};
  #pragma unroll
  for (int kk = 0; kk < F_IN; kk += 32) {
    v16bf ah, al, bh, bl;
    #pragma unroll
    for (int v = 0; v < 8; ++v) {
      const int kofs = akb + 2*v;
      bf2 a0 = split_bf16(xrow[kk + kofs]);
      bf2 a1 = split_bf16(xrow[kk + kofs + 1]);
      ah[2*v]   = a0.hi; al[2*v]   = a0.lo;
      ah[2*v+1] = a1.hi; al[2*v+1] = a1.lo;
      const int k0 = bkb + 2*v;
      bf2 b0 = split_bf16(wcol[(size_t)(kk + k0)     * HID]);
      bf2 b1v = split_bf16(wcol[(size_t)(kk + k0 + 1) * HID]);
      bh[2*v]   = b0.hi;  bl[2*v]   = b0.lo;
      bh[2*v+1] = b1v.hi; bl[2*v+1] = b1v.lo;
    }
    c = wmma_bf16(ah, bh, c);
    c = wmma_bf16(al, bh, c);
    c = wmma_bf16(ah, bl, c);
  }
  const int rbase = m0 + ((lane >> 4) * 8);
  const int col   = n0 + (lane & 15);
  #pragma unroll
  for (int v = 0; v < 8; ++v)
    h1[(size_t)(rbase + v) * HID + col] = c[v];
}

// ---------------- GEMM2: g[100000,16] = hact[100000,128] @ W2[128,16] -------------------
__global__ void __launch_bounds__(256)
k_gemm2(const float* __restrict__ hin, const float* __restrict__ W, float* __restrict__ g) {
  const int wave = threadIdx.x >> 5;
  const int lane = threadIdx.x & 31;
  const int mt = blockIdx.x * 8 + wave;          // wave-uniform tile index
  if (mt >= N_NODES / 16) return;                // uniform exit: WMMA EXEC stays all-1s
  const int m0 = mt * 16;

  const int am  = lane & 15;
  const int akb = (lane >> 4) * 16;
  const int bn  = lane & 15;
  const int bkb = (lane >> 4) * 16;

  const float* hrow = hin + (size_t)(m0 + am) * HID;
  const float* wcol = W + bn;

  v8f c = {};
  #pragma unroll
  for (int kk = 0; kk < HID; kk += 32) {
    v16bf ah, al, bh, bl;
    #pragma unroll
    for (int v = 0; v < 8; ++v) {
      const int kofs = akb + 2*v;
      bf2 a0 = split_bf16(hrow[kk + kofs]);
      bf2 a1 = split_bf16(hrow[kk + kofs + 1]);
      ah[2*v]   = a0.hi; al[2*v]   = a0.lo;
      ah[2*v+1] = a1.hi; al[2*v+1] = a1.lo;
      const int k0 = bkb + 2*v;
      bf2 b0 = split_bf16(wcol[(size_t)(kk + k0)     * NC]);
      bf2 b1v = split_bf16(wcol[(size_t)(kk + k0 + 1) * NC]);
      bh[2*v]   = b0.hi;  bl[2*v]   = b0.lo;
      bh[2*v+1] = b1v.hi; bl[2*v+1] = b1v.lo;
    }
    c = wmma_bf16(ah, bh, c);
    c = wmma_bf16(al, bh, c);
    c = wmma_bf16(ah, bl, c);
  }
  const int rbase = m0 + ((lane >> 4) * 8);
  const int col   = lane & 15;                   // n0 == 0
  #pragma unroll
  for (int v = 0; v < 8; ++v)
    g[(size_t)(rbase + v) * NC + col] = c[v];
}

// ---------------- per-node attention logits ---------------------------------------------
__global__ void k_alpha1(const float* __restrict__ h1, const float* __restrict__ as,
                         const float* __restrict__ ad, float* __restrict__ os,
                         float* __restrict__ od) {
  int t = blockIdx.x * blockDim.x + threadIdx.x;
  if (t >= N_NODES * HEADS) return;
  int n = t >> 2, k = t & 3;
  const float* hp = h1 + (size_t)n * HID + k * HDIM;
  const float* ap = as + k * HDIM;
  const float* bp = ad + k * HDIM;
  float s = 0.f, d = 0.f;
  #pragma unroll
  for (int i = 0; i < HDIM; ++i) { float v = hp[i]; s += v * ap[i]; d += v * bp[i]; }
  os[t] = s; od[t] = d;
}

__global__ void k_alpha2(const float* __restrict__ g, const float* __restrict__ as,
                         const float* __restrict__ ad, float* __restrict__ os,
                         float* __restrict__ od) {
  int n = blockIdx.x * blockDim.x + threadIdx.x;
  if (n >= N_NODES) return;
  const float* gp = g + (size_t)n * NC;
  float s = 0.f, d = 0.f;
  #pragma unroll
  for (int c = 0; c < NC; ++c) { float v = gp[c]; s += v * as[c]; d += v * ad[c]; }
  os[n] = s; od[n] = d;
}

__global__ void k_fill(float* p, long long n, float val) {
  long long t = (long long)blockIdx.x * blockDim.x + threadIdx.x;
  if (t < n) p[t] = val;
}

// ---------------- edge helpers ----------------------------------------------------------
__device__ inline void edge_sd(const long long* __restrict__ ei, int e, int& s, int& d) {
  if (e < N_EDGES) { s = (int)ei[e]; d = (int)ei[(size_t)N_EDGES + e]; }
  else             { s = d = e - N_EDGES; }     // self-loops
}

__global__ void k_edge_max1(const long long* __restrict__ ei, const float* __restrict__ as,
                            const float* __restrict__ ad, float* __restrict__ m1) {
  long long t = (long long)blockIdx.x * blockDim.x + threadIdx.x;
  if (t >= (long long)E_TOT * HEADS) return;
  int e = (int)(t >> 2), k = (int)(t & 3), s, d;
  edge_sd(ei, e, s, d);
  float v = as[s*HEADS + k] + ad[d*HEADS + k];
  v = v > 0.f ? v : NEG_SLOPE * v;
  atomicMaxF(&m1[d*HEADS + k], v);
}

__global__ void k_edge_exp1(const long long* __restrict__ ei, const float* __restrict__ as,
                            const float* __restrict__ ad, const float* __restrict__ m1,
                            float* __restrict__ ex, float* __restrict__ den) {
  long long t = (long long)blockIdx.x * blockDim.x + threadIdx.x;
  if (t >= (long long)E_TOT * HEADS) return;
  int e = (int)(t >> 2), k = (int)(t & 3), s, d;
  edge_sd(ei, e, s, d);
  float v = as[s*HEADS + k] + ad[d*HEADS + k];
  v = v > 0.f ? v : NEG_SLOPE * v;
  float x = expf(v - m1[d*HEADS + k]);
  ex[(size_t)e*HEADS + k] = x;
  atomicAdd(&den[d*HEADS + k], x);
}

__global__ void k_edge_agg1(const long long* __restrict__ ei, const float* __restrict__ h1,
                            const float* __restrict__ ex, const float* __restrict__ den,
                            float* __restrict__ agg) {
  long long t = (long long)blockIdx.x * blockDim.x + threadIdx.x;
  if (t >= (long long)E_TOT * HID) return;
  int e = (int)(t >> 7), f = (int)(t & 127), k = f >> 5, s, d;
  edge_sd(ei, e, s, d);
  float alpha = ex[(size_t)e*HEADS + k] / (den[d*HEADS + k] + 1e-16f);
  atomicAdd(&agg[(size_t)d*HID + f], h1[(size_t)s*HID + f] * alpha);
}

__global__ void k_elu_bias(float* __restrict__ a, const float* __restrict__ b1) {
  long long t = (long long)blockIdx.x * blockDim.x + threadIdx.x;
  if (t >= (long long)N_NODES * HID) return;
  float v = a[t] + b1[t & 127];
  a[t] = v > 0.f ? v : expm1f(v);
}

__global__ void k_edge_max2(const long long* __restrict__ ei, const float* __restrict__ as,
                            const float* __restrict__ ad, float* __restrict__ m2) {
  long long t = (long long)blockIdx.x * blockDim.x + threadIdx.x;
  if (t >= E_TOT) return;
  int e = (int)t, s, d;
  edge_sd(ei, e, s, d);
  float v = as[s] + ad[d];
  v = v > 0.f ? v : NEG_SLOPE * v;
  atomicMaxF(&m2[d], v);
}

__global__ void k_edge_exp2(const long long* __restrict__ ei, const float* __restrict__ as,
                            const float* __restrict__ ad, const float* __restrict__ m2,
                            float* __restrict__ ex, float* __restrict__ den) {
  long long t = (long long)blockIdx.x * blockDim.x + threadIdx.x;
  if (t >= E_TOT) return;
  int e = (int)t, s, d;
  edge_sd(ei, e, s, d);
  float v = as[s] + ad[d];
  v = v > 0.f ? v : NEG_SLOPE * v;
  float x = expf(v - m2[d]);
  ex[e] = x;
  atomicAdd(&den[d], x);
}

__global__ void k_edge_agg2(const long long* __restrict__ ei, const float* __restrict__ g,
                            const float* __restrict__ ex, const float* __restrict__ den,
                            float* __restrict__ agg) {
  long long t = (long long)blockIdx.x * blockDim.x + threadIdx.x;
  if (t >= (long long)E_TOT * NC) return;
  int e = (int)(t >> 4), f = (int)(t & 15), s, d;
  edge_sd(ei, e, s, d);
  float alpha = ex[e] / (den[d] + 1e-16f);
  atomicAdd(&agg[(size_t)d*NC + f], g[(size_t)s*NC + f] * alpha);
}

__global__ void k_final(const float* __restrict__ agg, const float* __restrict__ b2,
                        float* __restrict__ out) {
  int n = blockIdx.x * blockDim.x + threadIdx.x;
  if (n >= N_NODES) return;
  float v[NC]; float mx = -3.4e38f;
  #pragma unroll
  for (int c = 0; c < NC; ++c) { v[c] = agg[(size_t)n*NC + c] + b2[c]; mx = fmaxf(mx, v[c]); }
  float sum = 0.f;
  #pragma unroll
  for (int c = 0; c < NC; ++c) sum += expf(v[c] - mx);
  float l = mx + logf(sum);
  #pragma unroll
  for (int c = 0; c < NC; ++c) out[(size_t)n*NC + c] = v[c] - l;
}

// ---------------------------------------------------------------------------------------
extern "C" void kernel_launch(void* const* d_in, const int* in_sizes, int n_in,
                              void* d_out, int out_size, void* d_ws, size_t ws_size,
                              hipStream_t stream) {
  const float* x      = (const float*)d_in[0];
  const float* W1     = (const float*)d_in[1];
  const float* a_src1 = (const float*)d_in[2];
  const float* a_dst1 = (const float*)d_in[3];
  const float* b1     = (const float*)d_in[4];
  const float* W2     = (const float*)d_in[5];
  const float* a_src2 = (const float*)d_in[6];
  const float* a_dst2 = (const float*)d_in[7];
  const float* b2     = (const float*)d_in[8];
  const long long* ei = (const long long*)d_in[9];   // int64 [2, N_EDGES]
  float* out = (float*)d_out;

  char* ws = (char*)d_ws;
  size_t off = 0;
  auto alloc = [&](size_t bytes) -> void* {
    void* p = ws + off; off = (off + bytes + 255) & ~(size_t)255; return p;
  };
  float* h1     = (float*)alloc(sizeof(float)*(size_t)N_NODES*HID);
  float* agg1   = (float*)alloc(sizeof(float)*(size_t)N_NODES*HID);
  float* asrc1  = (float*)alloc(sizeof(float)*(size_t)N_NODES*HEADS);
  float* adst1  = (float*)alloc(sizeof(float)*(size_t)N_NODES*HEADS);
  float* m1     = (float*)alloc(sizeof(float)*(size_t)N_NODES*HEADS);
  float* den1   = (float*)alloc(sizeof(float)*(size_t)N_NODES*HEADS);
  float* ex1    = (float*)alloc(sizeof(float)*(size_t)E_TOT*HEADS);
  float* g2     = (float*)alloc(sizeof(float)*(size_t)N_NODES*NC);
  float* asrc2  = (float*)alloc(sizeof(float)*(size_t)N_NODES);
  float* adst2  = (float*)alloc(sizeof(float)*(size_t)N_NODES);
  float* m2     = (float*)alloc(sizeof(float)*(size_t)N_NODES);
  float* den2   = (float*)alloc(sizeof(float)*(size_t)N_NODES);
  float* ex2    = (float*)alloc(sizeof(float)*(size_t)E_TOT);
  float* agg2   = (float*)alloc(sizeof(float)*(size_t)N_NODES*NC);

  const int B = 256;
  auto nb = [](long long n, int b) { return (unsigned)((n + b - 1) / b); };

  // Layer 1
  k_gemm1<<<N_NODES/16, B, 0, stream>>>(x, W1, h1);
  k_alpha1<<<nb((long long)N_NODES*HEADS, B), B, 0, stream>>>(h1, a_src1, a_dst1, asrc1, adst1);
  k_fill<<<nb((long long)N_NODES*HEADS, B), B, 0, stream>>>(m1, (long long)N_NODES*HEADS,
                                                            -__builtin_inff());
  (void)hipMemsetAsync(den1, 0, sizeof(float)*(size_t)N_NODES*HEADS, stream);
  (void)hipMemsetAsync(agg1, 0, sizeof(float)*(size_t)N_NODES*HID, stream);
  k_edge_max1<<<nb((long long)E_TOT*HEADS, B), B, 0, stream>>>(ei, asrc1, adst1, m1);
  k_edge_exp1<<<nb((long long)E_TOT*HEADS, B), B, 0, stream>>>(ei, asrc1, adst1, m1, ex1, den1);
  k_edge_agg1<<<nb((long long)E_TOT*HID, B), B, 0, stream>>>(ei, h1, ex1, den1, agg1);
  k_elu_bias<<<nb((long long)N_NODES*HID, B), B, 0, stream>>>(agg1, b1);

  // Layer 2
  k_gemm2<<<nb(N_NODES/16, 8), B, 0, stream>>>(agg1, W2, g2);
  k_alpha2<<<nb(N_NODES, B), B, 0, stream>>>(g2, a_src2, a_dst2, asrc2, adst2);
  k_fill<<<nb(N_NODES, B), B, 0, stream>>>(m2, (long long)N_NODES, -__builtin_inff());
  (void)hipMemsetAsync(den2, 0, sizeof(float)*(size_t)N_NODES, stream);
  (void)hipMemsetAsync(agg2, 0, sizeof(float)*(size_t)N_NODES*NC, stream);
  k_edge_max2<<<nb(E_TOT, B), B, 0, stream>>>(ei, asrc2, adst2, m2);
  k_edge_exp2<<<nb(E_TOT, B), B, 0, stream>>>(ei, asrc2, adst2, m2, ex2, den2);
  k_edge_agg2<<<nb((long long)E_TOT*NC, B), B, 0, stream>>>(ei, g2, ex2, den2, agg2);
  k_final<<<nb(N_NODES, B), B, 0, stream>>>(agg2, b2, out);
}
